// ValueHead_1Trunk_44684839747696
// MI455X (gfx1250) — matrix operand order repaired and checked by reading, not compiled, not run on hardware
//
#include <hip/hip_runtime.h>

// Problem constants (from the reference)
#define CCH 256
#define HEADS 8
#define NPG 128
#define NGRAPH 2048
#define NN (NGRAPH * NPG)
#define EPSV 1e-5f
#define ATT_SCALE 0.17677669529663687f /* 1/sqrt(32) */

// LDS row stride (bf16 elems) for W1 staging: 264*2B = 528B -> 16B aligned,
// stride of 132 dwords spreads 16 rows across distinct bank quads.
#define W1S 264

#if defined(__gfx1250__) && __has_builtin(__builtin_amdgcn_global_load_async_to_lds_b128)
#define HAVE_ASYNC_LDS 1
#else
#define HAVE_ASYNC_LDS 0
#endif

typedef __bf16 v16bf __attribute__((ext_vector_type(16)));
typedef float v8f __attribute__((ext_vector_type(8)));

// Types matching the async-to-LDS builtin signature:
//   param0: 'int __vector(4) __device__ *'  (global, AS1)
//   param1: LDS destination (AS3)
typedef int b128v __attribute__((vector_size(16)));
typedef __attribute__((address_space(1))) b128v* gas_b128p;
typedef __attribute__((address_space(3))) b128v* las_b128p;

union Frag {
  v16bf v;
  unsigned short h[16];
  uint4 q[2];
};

__device__ __forceinline__ unsigned short f2bf(float f) {
  unsigned u = __builtin_bit_cast(unsigned, f);
  u += 0x7fffu + ((u >> 16) & 1u); // round-to-nearest-even
  return (unsigned short)(u >> 16);
}
__device__ __forceinline__ float bf2f(unsigned short h) {
  unsigned u = ((unsigned)h) << 16;
  return __builtin_bit_cast(float, u);
}
__device__ __forceinline__ unsigned pack2(unsigned short a, unsigned short b) {
  return (unsigned)a | ((unsigned)b << 16);
}

// ---------------------------------------------------------------------------
// k0: convert W1 -> bf16 into workspace, zero the stat accumulators.
// ---------------------------------------------------------------------------
__global__ __launch_bounds__(256) void k0_prep(const float* __restrict__ W1,
                                               unsigned short* __restrict__ w1bf,
                                               float* __restrict__ stats) {
  int i = blockIdx.x * 256 + threadIdx.x; // 65536 threads
  if (i < 512) stats[i] = 0.f;
  w1bf[i] = f2bf(W1[i]);
}

// ---------------------------------------------------------------------------
// k1: h1 = x @ W1^T + b1 via v_wmma_f32_16x16x32_bf16.
//     One WG (8 waves) per 128 rows; each wave owns a 16-row strip and all
//     16 column tiles. W1 staged through LDS in 4 stages of 64 rows using
//     GLOBAL_LOAD_ASYNC_TO_LDS_B128 (ASYNCcnt) when available.
//     Outputs h1 transposed (channel-major, bf16) + per-channel sum/sumsq.
// ---------------------------------------------------------------------------
__global__ __launch_bounds__(256) void k1_gemm1(
    const float* __restrict__ x, const float* __restrict__ b1,
    const unsigned short* __restrict__ w1bf,
    unsigned short* __restrict__ h1t,
    float* __restrict__ gsum, float* __restrict__ gsq) {
  __shared__ __align__(16) unsigned short w1l[64 * W1S];
  __shared__ float lsum[CCH];
  __shared__ float lsq[CCH];

  const int tid = threadIdx.x;
  const int wave = tid >> 5;
  const int lane = tid & 31;
  const int lrow = lane & 15;
  const int hi = lane >> 4; // 0: low lane-half, 1: high lane-half

  lsum[tid] = 0.f;
  lsq[tid] = 0.f;

  const int m0 = blockIdx.x * 128 + wave * 16;
  const int arow = m0 + lrow;
  const int kb = hi * 8; // A-matrix K-base per ISA 16-bit A 16x32 layout

  // Load x strip, convert to bf16 A-fragments held in registers (8 x v16bf).
  Frag A[8];
  const float* xr = x + (size_t)arow * CCH;
#pragma unroll
  for (int ks = 0; ks < 8; ++ks) {
    const int k0 = ks * 32 + kb;
    float4 a0 = *(const float4*)(xr + k0);
    float4 a1 = *(const float4*)(xr + k0 + 4);
    float4 c0 = *(const float4*)(xr + k0 + 16);
    float4 c1 = *(const float4*)(xr + k0 + 20);
    A[ks].q[0].x = pack2(f2bf(a0.x), f2bf(a0.y));
    A[ks].q[0].y = pack2(f2bf(a0.z), f2bf(a0.w));
    A[ks].q[0].z = pack2(f2bf(a1.x), f2bf(a1.y));
    A[ks].q[0].w = pack2(f2bf(a1.z), f2bf(a1.w));
    A[ks].q[1].x = pack2(f2bf(c0.x), f2bf(c0.y));
    A[ks].q[1].y = pack2(f2bf(c0.z), f2bf(c0.w));
    A[ks].q[1].z = pack2(f2bf(c1.x), f2bf(c1.y));
    A[ks].q[1].w = pack2(f2bf(c1.z), f2bf(c1.w));
  }

  const int koffB = hi * 8; // B-matrix K-base per ISA 16-bit B layout

  for (int stage = 0; stage < 4; ++stage) {
    __syncthreads(); // protect previous stage's consumers before overwrite
    // Stage W1 rows [stage*64, stage*64+64) into LDS.
#if HAVE_ASYNC_LDS
    for (int i = tid; i < 64 * 32; i += 256) {
      const int row = i >> 5;
      const int col = i & 31;
      const uint4* gp = ((const uint4*)w1bf) + ((size_t)(stage * 64 + row) << 5) + col;
      __builtin_amdgcn_global_load_async_to_lds_b128(
          (gas_b128p)(b128v*)gp,
          (las_b128p)(b128v*)&w1l[row * W1S + col * 8],
          0, 0);
    }
#if __has_builtin(__builtin_amdgcn_s_wait_asynccnt)
    __builtin_amdgcn_s_wait_asynccnt(0);
#else
    asm volatile("s_wait_asynccnt 0x0" ::: "memory");
#endif
#else
    for (int i = tid; i < 64 * 32; i += 256) {
      const int row = i >> 5;
      const int col = i & 31;
      uint4 d = ((const uint4*)w1bf)[((size_t)(stage * 64 + row) << 5) + col];
      *(uint4*)&w1l[row * W1S + col * 8] = d;
    }
#endif
    __syncthreads();

#pragma unroll
    for (int ct4 = 0; ct4 < 4; ++ct4) {
      const int ct = stage * 4 + ct4;
      const int browl = ct4 * 16 + lrow; // B column n = W1 row (ct*16 + n)

      // Batch all 8 B fragments (16 ds_load_b128) before the WMMA chain so
      // the scheduler can overlap LDS returns with matrix issue.
      Frag B[8];
#pragma unroll
      for (int ks = 0; ks < 8; ++ks) {
        const unsigned short* bp = &w1l[browl * W1S + ks * 32 + koffB];
        B[ks].q[0] = *(const uint4*)(bp);
        B[ks].q[1] = *(const uint4*)(bp + 16);
      }
      v8f acc = {};
#pragma unroll
      for (int ks = 0; ks < 8; ++ks) {
        acc = __builtin_amdgcn_wmma_f32_16x16x32_bf16(
            false, A[ks].v, false, B[ks].v, (short)0, acc, false, false);
      }
      // D layout: vgpr r -> M = r + 8*hi, N = lane&15. Same channel c per lane.
      const int c = ct * 16 + lrow;
      const float bias = b1[c];
      float s1 = 0.f, s2 = 0.f;
      unsigned short ph[8];
#pragma unroll
      for (int r = 0; r < 8; ++r) {
        float v = acc[r] + bias;
        s1 += v;
        s2 += v * v;
        ph[r] = f2bf(v);
      }
      atomicAdd(&lsum[c], s1);
      atomicAdd(&lsq[c], s2);
      uint4 pk;
      pk.x = pack2(ph[0], ph[1]);
      pk.y = pack2(ph[2], ph[3]);
      pk.z = pack2(ph[4], ph[5]);
      pk.w = pack2(ph[6], ph[7]);
      // channel-major store: 8 consecutive nodes -> one 16B store
      *(uint4*)(h1t + (size_t)c * NN + m0 + hi * 8) = pk;
    }
  }
  __syncthreads();
  atomicAdd(&gsum[tid], lsum[tid]);
  atomicAdd(&gsq[tid], lsq[tid]);
}

// ---------------------------------------------------------------------------
// k2: one WG. BN affine (a,s), folded projection P=[M|G] (256x16), constants.
// ---------------------------------------------------------------------------
__global__ __launch_bounds__(256) void k2_pre(
    const float* __restrict__ gsum, const float* __restrict__ gsq,
    const float* __restrict__ gamma, const float* __restrict__ beta,
    const float* __restrict__ W2, const float* __restrict__ b2,
    const float* __restrict__ Wq, const float* __restrict__ bq,
    const float* __restrict__ Wk, const float* __restrict__ bk,
    const float* __restrict__ Wv, const float* __restrict__ bv,
    const float* __restrict__ Wo, const float* __restrict__ bo,
    const float* __restrict__ Wr, const float* __restrict__ br,
    float* __restrict__ aff, float* __restrict__ Pmat,
    float* __restrict__ consts) {
  __shared__ float qraw[CCH];
  __shared__ float wr2s[CCH];
  __shared__ float wkq[CCH][HEADS];
  __shared__ float wvr[HEADS][CCH];
  __shared__ float bwv[CCH];
  const int t = threadIdx.x;

  const float inv_n = 1.0f / (float)NN;
  float mu = gsum[t] * inv_n;
  float var = gsq[t] * inv_n - mu * mu;
  float rs = rsqrtf(var + EPSV);
  float a = gamma[t] * rs;
  aff[t] = a;
  aff[CCH + t] = beta[t] - mu * a;

  // qh (q = ones): qraw[j] = sum_k Wq[j,k] + bq[j]
  float q = bq[t];
  for (int k = 0; k < CCH; ++k) q += Wq[t * CCH + k];
  qraw[t] = q;
  __syncthreads();

  { // wkq[c,h] = scale * sum_d qraw[h*32+d] * Wk[h*32+d, c]
    float acc[HEADS];
#pragma unroll
    for (int h = 0; h < HEADS; ++h) acc[h] = 0.f;
#pragma unroll
    for (int h = 0; h < HEADS; ++h)
      for (int d = 0; d < 32; ++d) {
        int hd = h * 32 + d;
        acc[h] += qraw[hd] * Wk[hd * CCH + t];
      }
#pragma unroll
    for (int h = 0; h < HEADS; ++h) wkq[t][h] = acc[h] * ATT_SCALE;
  }
  { // wr2[hd] = sum_j Wr[0,j] * Wo[j, hd]
    float w = 0.f;
    for (int j = 0; j < CCH; ++j) w += Wr[j] * Wo[j * CCH + t];
    wr2s[t] = w;
  }
  __syncthreads();
  { // wvr[h,c] = sum_d wr2[h*32+d] * Wv[h*32+d, c]
    float acc[HEADS];
#pragma unroll
    for (int h = 0; h < HEADS; ++h) acc[h] = 0.f;
#pragma unroll
    for (int h = 0; h < HEADS; ++h)
      for (int d = 0; d < 32; ++d) {
        int hd = h * 32 + d;
        acc[h] += wr2s[hd] * Wv[hd * CCH + t];
      }
#pragma unroll
    for (int h = 0; h < HEADS; ++h) wvr[h][t] = acc[h];
  }
  { // bwv[hd] = b2 . Wv[hd,:]
    float bb = 0.f;
    for (int c = 0; c < CCH; ++c) bb += b2[c] * Wv[t * CCH + c];
    bwv[t] = bb;
  }
  __syncthreads();
  { // P[c',0..7] = M = W2^T wkq ; P[c',8..15] = G = (wvr W2)^T
    float M[HEADS], G[HEADS];
#pragma unroll
    for (int h = 0; h < HEADS; ++h) { M[h] = 0.f; G[h] = 0.f; }
    for (int c = 0; c < CCH; ++c) {
      float w2 = W2[c * CCH + t];
#pragma unroll
      for (int h = 0; h < HEADS; ++h) {
        M[h] += w2 * wkq[c][h];
        G[h] += w2 * wvr[h][c];
      }
    }
#pragma unroll
    for (int h = 0; h < HEADS; ++h) {
      Pmat[t * 16 + h] = M[h];
      Pmat[t * 16 + 8 + h] = G[h];
    }
  }
  if (t < HEADS) { // logit constants
    float lc = 0.f;
    for (int d = 0; d < 32; ++d) lc += qraw[t * 32 + d] * bk[t * 32 + d];
    lc *= ATT_SCALE;
    for (int c = 0; c < CCH; ++c) lc += b2[c] * wkq[c][t];
    consts[t] = lc;
  }
  if (t == HEADS) { // scalar constant K
    float kc = br[0];
    for (int j = 0; j < CCH; ++j) kc += Wr[j] * bo[j];
    for (int hd = 0; hd < CCH; ++hd) kc += wr2s[hd] * (bwv[hd] + bv[hd]);
    consts[8] = kc;
  }
}

// ---------------------------------------------------------------------------
// k3: one WG (128 threads, 4 waves) per graph. Per-node: BN affine + ReLU,
//     256->16 projection (logits | u), per-head softmax, tanh output.
// ---------------------------------------------------------------------------
__global__ __launch_bounds__(128) void k3_attn(
    const unsigned short* __restrict__ h1t,
    const float* __restrict__ Pmat, const float* __restrict__ aff,
    const float* __restrict__ consts, float* __restrict__ out) {
  __shared__ __align__(16) float Pl[CCH * 16];
  __shared__ float al[CCH];
  __shared__ float sl[CCH];
  __shared__ float redm[4], redd[4], redu[4];

  const int tid = threadIdx.x;
  const int g = blockIdx.x;
  const int lane = tid & 31;
  const int wid = tid >> 5;

  for (int i = tid; i < CCH * 16; i += 128) Pl[i] = Pmat[i];
  for (int i = tid; i < CCH; i += 128) {
    al[i] = aff[i];
    sl[i] = aff[CCH + i];
  }
  float lc[9];
#pragma unroll
  for (int i = 0; i < 9; ++i) lc[i] = consts[i];
  __syncthreads();

  const int n = g * NPG + tid;
  float y[16];
#pragma unroll
  for (int j = 0; j < 16; ++j) y[j] = 0.f;

  for (int c = 0; c < CCH; ++c) {
    if (c + 16 < CCH)
      __builtin_prefetch(h1t + (size_t)(c + 16) * NN + n, 0, 0);
    float h1v = bf2f(h1t[(size_t)c * NN + n]); // coalesced: 128 x 2B
    float hn = fmaxf(fmaf(al[c], h1v, sl[c]), 0.f);
    const float4 p0 = *(const float4*)&Pl[c * 16 + 0];
    const float4 p1 = *(const float4*)&Pl[c * 16 + 4];
    const float4 p2 = *(const float4*)&Pl[c * 16 + 8];
    const float4 p3 = *(const float4*)&Pl[c * 16 + 12];
    y[0] = fmaf(hn, p0.x, y[0]);   y[1] = fmaf(hn, p0.y, y[1]);
    y[2] = fmaf(hn, p0.z, y[2]);   y[3] = fmaf(hn, p0.w, y[3]);
    y[4] = fmaf(hn, p1.x, y[4]);   y[5] = fmaf(hn, p1.y, y[5]);
    y[6] = fmaf(hn, p1.z, y[6]);   y[7] = fmaf(hn, p1.w, y[7]);
    y[8] = fmaf(hn, p2.x, y[8]);   y[9] = fmaf(hn, p2.y, y[9]);
    y[10] = fmaf(hn, p2.z, y[10]); y[11] = fmaf(hn, p2.w, y[11]);
    y[12] = fmaf(hn, p3.x, y[12]); y[13] = fmaf(hn, p3.y, y[13]);
    y[14] = fmaf(hn, p3.z, y[14]); y[15] = fmaf(hn, p3.w, y[15]);
  }

  float total = lc[8];
#pragma unroll
  for (int h = 0; h < HEADS; ++h) {
    float l = y[h] + lc[h];
    float m = l;
#pragma unroll
    for (int off = 16; off > 0; off >>= 1) m = fmaxf(m, __shfl_xor(m, off));
    if (lane == 0) redm[wid] = m;
    __syncthreads();
    m = fmaxf(fmaxf(redm[0], redm[1]), fmaxf(redm[2], redm[3]));
    float e = __expf(l - m);
    float eu = e * y[8 + h];
    float se = e, seu = eu;
#pragma unroll
    for (int off = 16; off > 0; off >>= 1) {
      se += __shfl_xor(se, off);
      seu += __shfl_xor(seu, off);
    }
    if (lane == 0) {
      redd[wid] = se;
      redu[wid] = seu;
    }
    __syncthreads();
    float denom = redd[0] + redd[1] + redd[2] + redd[3];
    float wsum = redu[0] + redu[1] + redu[2] + redu[3];
    total += wsum / denom;
    __syncthreads();
  }
  if (tid == 0) out[g] = tanhf(total);
}

// ---------------------------------------------------------------------------
extern "C" void kernel_launch(void* const* d_in, const int* in_sizes, int n_in,
                              void* d_out, int out_size, void* d_ws,
                              size_t ws_size, hipStream_t stream) {
  const float* x = (const float*)d_in[0];
  // d_in[1] = batch (unused: segments are contiguous blocks of 128)
  // d_in[2] = num_graphs (compile-time constant 2048)
  const float* W1 = (const float*)d_in[3];
  const float* b1 = (const float*)d_in[4];
  const float* gamma = (const float*)d_in[5];
  const float* beta = (const float*)d_in[6];
  const float* W2 = (const float*)d_in[7];
  const float* b2 = (const float*)d_in[8];
  const float* Wq = (const float*)d_in[9];
  const float* bq = (const float*)d_in[10];
  const float* Wk = (const float*)d_in[11];
  const float* bk = (const float*)d_in[12];
  const float* Wv = (const float*)d_in[13];
  const float* bv = (const float*)d_in[14];
  const float* Wo = (const float*)d_in[15];
  const float* bo = (const float*)d_in[16];
  const float* Wr = (const float*)d_in[17];
  const float* br = (const float*)d_in[18];

  float* ws = (float*)d_ws;
  float* stats = ws;          // gsum[256] | gsq[256]
  float* gsum = ws;
  float* gsq = ws + 256;
  float* consts = ws + 512;   // lc[8], Kc
  float* aff = ws + 528;      // a[256], s[256]
  float* Pmat = ws + 1040;    // 256*16
  unsigned short* w1bf = (unsigned short*)(ws + 5136);         // 65536 bf16
  unsigned short* h1t = (unsigned short*)(ws + 5136 + 32768);  // NN*CCH bf16

  hipLaunchKernelGGL(k0_prep, dim3(256), dim3(256), 0, stream, W1, w1bf, stats);
  hipLaunchKernelGGL(k1_gemm1, dim3(NGRAPH), dim3(256), 0, stream, x, b1, w1bf,
                     h1t, gsum, gsq);
  hipLaunchKernelGGL(k2_pre, dim3(1), dim3(256), 0, stream, gsum, gsq, gamma,
                     beta, W2, b2, Wq, bq, Wk, bk, Wv, bv, Wo, bo, Wr, br, aff,
                     Pmat, consts);
  hipLaunchKernelGGL(k3_attn, dim3(NGRAPH), dim3(128), 0, stream, h1t, Pmat,
                     aff, consts, (float*)d_out);
  (void)in_sizes; (void)n_in; (void)out_size; (void)ws_size;
}
